// GAT_BiLSTM_Attn_Temporal_44444321579695
// MI455X (gfx1250) — matrix operand order repaired and checked by reading, not compiled, not run on hardware
//
#include <hip/hip_runtime.h>
#include <hip/hip_bf16.h>
#include <math.h>

#define N_NODES 20000
#define N_EDGES 320000
#define ETOT    (N_EDGES + N_NODES)   // edges + self loops
#define BSZ     128
#define TT      200
#define EMB     128
#define GH      64
#define HEADS   4
#define LH      128
#define NSP     64
#define K1PAD   160                    // 134 padded to multiple of 32
#define HC1     256                    // HEADS*GH
#define BT      (BSZ*TT)               // 25600

typedef _Float16 v16h __attribute__((ext_vector_type(16)));
typedef _Float16 v8h  __attribute__((ext_vector_type(8)));
typedef float    v8f  __attribute__((ext_vector_type(8)));

__device__ __forceinline__ float sigf(float x) { return 1.f / (1.f + expf(-x)); }

// ---------------------------------------------------------------------------
// Register-blocked WMMA GEMM: C[M,N](f32) = A[M,K](f16) * Bt[N,K](f16)^T
// Each wave owns an (MB*16) x (NB*16) macro-tile: MB A-fragments + NB
// B-fragments per 32-wide K-step feed MB*NB v_wmma_f32_16x16x32_f16 ops,
// raising arithmetic intensity ~3x over a single-tile wave.
// ---------------------------------------------------------------------------
template <int MB, int NB>
__global__ void gemm_wmma_blk(const _Float16* __restrict__ A, int lda,
                              const _Float16* __restrict__ Bt, int ldb,
                              float* __restrict__ C, int ldc,
                              const float* __restrict__ bias,
                              int Mgrp, int Ngrp, int Ksteps) {
  int wave = blockIdx.x * (blockDim.x >> 5) + (threadIdx.x >> 5);
  if (wave >= Mgrp * Ngrp) return;              // wave-uniform
  int mg = wave / Ngrp, ng = wave - mg * Ngrp;
  int lane = threadIdx.x & 31;
  int lr = lane & 15;          // row-of-tile (A) / col-of-tile (B)
  int kh = lane >> 4;          // k-half select
  const _Float16* ap[MB];
  const _Float16* bp[NB];
#pragma unroll
  for (int i = 0; i < MB; ++i)
    ap[i] = A + (size_t)((mg * MB + i) * 16 + lr) * lda + kh * 16;
#pragma unroll
  for (int j = 0; j < NB; ++j)
    bp[j] = Bt + (size_t)((ng * NB + j) * 16 + lr) * ldb + kh * 16;

  v8f acc[MB][NB] = {};
  for (int ks = 0; ks < Ksteps; ++ks) {
    v16h af[MB], bf[NB];
#pragma unroll
    for (int i = 0; i < MB; ++i) {
      union { v16h v; v8h h[2]; } u;
      u.h[0] = *(const v8h*)(ap[i]);        // K = 16*kh + 0..7
      u.h[1] = *(const v8h*)(ap[i] + 8);    // K = 16*kh + 8..15
      af[i] = u.v;
      ap[i] += 32;
    }
#pragma unroll
    for (int j = 0; j < NB; ++j) {
      bf[j] = *(const v16h*)(bp[j]);        // K = 16*kh + 0..15 contiguous
      bp[j] += 32;
    }
#pragma unroll
    for (int i = 0; i < MB; ++i)
#pragma unroll
      for (int j = 0; j < NB; ++j)
        acc[i][j] = __builtin_amdgcn_wmma_f32_16x16x32_f16(
            false, af[i], false, bf[j], (short)0, acc[i][j], false, false);
  }

#pragma unroll
  for (int i = 0; i < MB; ++i) {
    int rb = (mg * MB + i) * 16 + kh * 8;
#pragma unroll
    for (int j = 0; j < NB; ++j) {
      int col = (ng * NB + j) * 16 + lr;
      float bv = bias ? bias[col] : 0.f;
      size_t base = (size_t)rb * ldc + col;
#pragma unroll
      for (int r = 0; r < 8; ++r) C[base + (size_t)r * ldc] = acc[i][j][r] + bv;
    }
  }
}

// ---------------------------------------------------------------------------
// Fused LSTM step: one block = 4 waves = 4 gates for one (16 batch,16 hidden)
// tile. WMMA for h_prev @ W_hh^T, LDS exchange, then gate math + state update.
// ---------------------------------------------------------------------------
__global__ void lstm_step(const _Float16* __restrict__ hprev, // [128][128] f16
                          const _Float16* __restrict__ whh,   // [512][128] f16
                          const float* __restrict__ xg,       // [BT][512] f32
                          const float* __restrict__ bias,     // [512]
                          float* __restrict__ cst,            // [128][128]
                          _Float16* __restrict__ hnext,       // [128][128]
                          float* __restrict__ outbuf,         // [B][T][128]
                          int t) {
  __shared__ float gl[4][256];
  int g = threadIdx.x >> 5;            // gate index (i,f,g,o)
  int lane = threadIdx.x & 31;
  int mt = blockIdx.x, lt = blockIdx.y;
  int lr = lane & 15, kh = lane >> 4;
  const _Float16* ap = hprev + (size_t)(mt * 16 + lr) * 128 + kh * 16;
  const _Float16* bp = whh + (size_t)(g * 128 + lt * 16 + lr) * 128 + kh * 16;
  v8f acc = {};
#pragma unroll
  for (int ks = 0; ks < 4; ++ks) {
    union { v16h v; v8h h[2]; } af;
    af.h[0] = *(const v8h*)(ap + ks * 32);
    af.h[1] = *(const v8h*)(ap + ks * 32 + 8);
    v16h bf = *(const v16h*)(bp + ks * 32);
    acc = __builtin_amdgcn_wmma_f32_16x16x32_f16(
        false, af.v, false, bf, (short)0, acc, false, false);
  }
#pragma unroll
  for (int j = 0; j < 8; ++j) gl[g][(kh * 8 + j) * 16 + lr] = acc[j];
  __syncthreads();
#pragma unroll
  for (int r = 0; r < 2; ++r) {
    int elem = threadIdx.x + r * 128;   // 256 tile elements, 128 threads
    int mi = elem >> 4, li = elem & 15;
    int b = mt * 16 + mi, l = lt * 16 + li;
    size_t xrow = ((size_t)b * TT + t) * 512;
    float gi = gl[0][elem] + xg[xrow + 0   + l] + bias[0   + l];
    float gf = gl[1][elem] + xg[xrow + 128 + l] + bias[128 + l];
    float gg = gl[2][elem] + xg[xrow + 256 + l] + bias[256 + l];
    float go = gl[3][elem] + xg[xrow + 384 + l] + bias[384 + l];
    float c_old = cst[b * 128 + l];
    float c_new = sigf(gf) * c_old + sigf(gi) * tanhf(gg);
    float h = sigf(go) * tanhf(c_new);
    cst[b * 128 + l] = c_new;
    hnext[b * 128 + l] = (_Float16)h;
    outbuf[((size_t)b * TT + t) * 128 + l] = h;
  }
}

// ---------------------------------------------------------------------------
// Prep / conversion kernels
// ---------------------------------------------------------------------------
__global__ void k_prep_feat(const float* __restrict__ xc, const float* __restrict__ ne,
                            const float* __restrict__ tf, _Float16* __restrict__ out) {
  int idx = blockIdx.x * blockDim.x + threadIdx.x;
  if (idx >= N_NODES * K1PAD) return;
  int n = idx / K1PAD, c = idx - n * K1PAD;
  float v = 0.f;
  if (c < 2)        v = xc[n * 2 + c];
  else if (c < 130) v = ne[n * EMB + (c - 2)];
  else if (c < 134) v = tf[n * 4 + (c - 130)];
  out[idx] = (_Float16)v;
}

// src[K,O] f32 -> dst[O,Kpad] f16 (transpose + zero pad)
__global__ void k_transpose_cvt(const float* __restrict__ src, _Float16* __restrict__ dst,
                                int K, int O, int Kpad) {
  long long idx = (long long)blockIdx.x * blockDim.x + threadIdx.x;
  if (idx >= (long long)O * Kpad) return;
  int o = (int)(idx / Kpad), k = (int)(idx - (long long)o * Kpad);
  dst[idx] = (k < K) ? (_Float16)src[(size_t)k * O + o] : (_Float16)0.f;
}

__global__ void k_cvt(const float* __restrict__ src, _Float16* __restrict__ dst, int n) {
  int idx = blockIdx.x * blockDim.x + threadIdx.x;
  if (idx < n) dst[idx] = (_Float16)src[idx];
}

// ---------------------------------------------------------------------------
// GAT kernels
// ---------------------------------------------------------------------------
__global__ void k_asad(const float* __restrict__ H, const float* __restrict__ att_s,
                       const float* __restrict__ att_d, float* __restrict__ as_,
                       float* __restrict__ ad_, int heads, int ch) {
  int idx = blockIdx.x * blockDim.x + threadIdx.x;
  if (idx >= N_NODES * heads) return;
  int n = idx / heads, h = idx - n * heads;
  const float* p = H + (size_t)n * (heads * ch) + h * ch;
  const float* s = att_s + h * ch;
  const float* d = att_d + h * ch;
  float vs = 0.f, vd = 0.f;
  for (int c = 0; c < ch; ++c) { float v = p[c]; vs += v * s[c]; vd += v * d[c]; }
  as_[idx] = vs; ad_[idx] = vd;
}

__global__ void k_edge_w(const float* __restrict__ as_, const float* __restrict__ ad_,
                         const int* __restrict__ ei, float* __restrict__ w,
                         float* __restrict__ denom, int heads) {
  long long idx = (long long)blockIdx.x * blockDim.x + threadIdx.x;
  if (idx >= (long long)ETOT * heads) return;
  int e = (int)(idx / heads), h = (int)(idx - (long long)e * heads);
  int src = (e < N_EDGES) ? ei[e] : e - N_EDGES;
  int dst = (e < N_EDGES) ? ei[N_EDGES + e] : e - N_EDGES;
  float x = as_[src * heads + h] + ad_[dst * heads + h];
  x = x > 0.f ? x : 0.2f * x;                 // leaky_relu(0.2)
  float wv = expf(x);                          // softmax shift-invariant: skip max
  w[idx] = wv;
  atomicAdd(&denom[dst * heads + h], wv);
}

__global__ void k_edge_agg(const float* __restrict__ H, const float* __restrict__ w,
                           const float* __restrict__ denom, const int* __restrict__ ei,
                           float* __restrict__ out, int heads, int ch) {
  long long idx = (long long)blockIdx.x * blockDim.x + threadIdx.x;
  int HC = heads * ch;
  if (idx >= (long long)ETOT * HC) return;
  int e = (int)(idx / HC), hc = (int)(idx - (long long)e * HC);
  int h = hc / ch;
  int src = (e < N_EDGES) ? ei[e] : e - N_EDGES;
  int dst = (e < N_EDGES) ? ei[N_EDGES + e] : e - N_EDGES;
  float alpha = w[(size_t)e * heads + h] / denom[dst * heads + h];
  atomicAdd(&out[(size_t)dst * HC + hc], H[(size_t)src * HC + hc] * alpha);
}

__global__ void k_bias_elu_half(const float* __restrict__ z, const float* __restrict__ bias,
                                _Float16* __restrict__ out, int HC) {
  long long idx = (long long)blockIdx.x * blockDim.x + threadIdx.x;
  if (idx >= (long long)N_NODES * HC) return;
  int c = (int)(idx % HC);
  float v = z[idx] + bias[c];
  v = v > 0.f ? v : (expf(v) - 1.f);           // elu
  out[idx] = (_Float16)v;
}

// gather z[seq] (+bias) into forward & length-reversed sequence buffers (f16)
__global__ void k_gather(const float* __restrict__ z, const float* __restrict__ bias,
                         const int* __restrict__ seq, const int* __restrict__ lengths,
                         _Float16* __restrict__ semb, _Float16* __restrict__ xrev) {
  long long idx = (long long)blockIdx.x * blockDim.x + threadIdx.x;
  if (idx >= (long long)BT * EMB) return;
  int b = (int)(idx / (TT * EMB));
  int r = (int)(idx - (long long)b * TT * EMB);
  int t = r / EMB, c = r - t * EMB;
  int node = seq[b * TT + t];
  semb[idx] = (_Float16)(z[(size_t)node * EMB + c] + bias[c]);
  int len = lengths[b];
  int rt = len - 1 - t; rt = rt < 0 ? 0 : (rt > TT - 1 ? TT - 1 : rt);
  int noder = seq[b * TT + rt];
  xrev[idx] = (_Float16)(z[(size_t)noder * EMB + c] + bias[c]);
}

// ---------------------------------------------------------------------------
// Masked attention pooling over time (both pools), ctx emitted as f16 [B,256]
// ---------------------------------------------------------------------------
__global__ void k_attn(const float* __restrict__ outf, const float* __restrict__ outrev,
                       const int* __restrict__ lengths,
                       const float* __restrict__ wn, const float* __restrict__ bn,
                       const float* __restrict__ wsp, const float* __restrict__ bsp,
                       _Float16* __restrict__ ctxn, _Float16* __restrict__ ctxs) {
  int b = blockIdx.x, tid = threadIdx.x;
  __shared__ float scn[256], scs[256], red[4];
  int len = lengths[b];
  if (tid < TT) {
    int rt = len - 1 - tid; rt = rt < 0 ? 0 : (rt > TT - 1 ? TT - 1 : rt);
    const float* pf = outf + ((size_t)b * TT + tid) * 128;
    const float* pb = outrev + ((size_t)b * TT + rt) * 128;
    float sn = bn[0], ss = bsp[0];
    for (int h = 0; h < 128; ++h) { float v = pf[h]; sn += v * wn[h];       ss += v * wsp[h]; }
    for (int h = 0; h < 128; ++h) { float v = pb[h]; sn += v * wn[128 + h]; ss += v * wsp[128 + h]; }
    bool m = tid < len;
    scn[tid] = m ? sn : -1e30f;
    scs[tid] = m ? ss : -1e30f;
  } else { scn[tid] = -1e30f; scs[tid] = -1e30f; }
  __syncthreads();
  if (tid == 0) { float mx = -1e30f; for (int t = 0; t < TT; ++t) mx = fmaxf(mx, scn[t]); red[0] = mx; }
  if (tid == 1) { float mx = -1e30f; for (int t = 0; t < TT; ++t) mx = fmaxf(mx, scs[t]); red[2] = mx; }
  __syncthreads();
  if (tid < TT) { scn[tid] = expf(scn[tid] - red[0]); scs[tid] = expf(scs[tid] - red[2]); }
  else          { scn[tid] = 0.f; scs[tid] = 0.f; }
  __syncthreads();
  if (tid == 0) { float s = 0.f; for (int t = 0; t < TT; ++t) s += scn[t]; red[1] = s; }
  if (tid == 1) { float s = 0.f; for (int t = 0; t < TT; ++t) s += scs[t]; red[3] = s; }
  __syncthreads();
  float invn = 1.f / red[1], invs = 1.f / red[3];
  float an = 0.f, asp = 0.f;
  for (int t = 0; t < TT; ++t) {
    float lv;
    if (tid < 128) lv = outf[((size_t)b * TT + t) * 128 + tid];
    else {
      int rt = len - 1 - t; rt = rt < 0 ? 0 : (rt > TT - 1 ? TT - 1 : rt);
      lv = outrev[((size_t)b * TT + rt) * 128 + (tid - 128)];
    }
    an += scn[t] * lv; asp += scs[t] * lv;
  }
  ctxn[b * 256 + tid] = (_Float16)(an * invn);
  ctxs[b * 256 + tid] = (_Float16)(asp * invs);
}

// ---------------------------------------------------------------------------
extern "C" void kernel_launch(void* const* d_in, const int* in_sizes, int n_in,
                              void* d_out, int out_size, void* d_ws, size_t ws_size,
                              hipStream_t stream) {
  (void)in_sizes; (void)n_in; (void)out_size; (void)ws_size;
  const float* x_coords  = (const float*)d_in[0];
  const float* temporal  = (const float*)d_in[1];
  const float* node_emb  = (const float*)d_in[2];
  const float* gat1_W    = (const float*)d_in[3];
  const float* g1_as     = (const float*)d_in[4];
  const float* g1_ad     = (const float*)d_in[5];
  const float* g1_bias   = (const float*)d_in[6];
  const float* gat2_W    = (const float*)d_in[7];
  const float* g2_as     = (const float*)d_in[8];
  const float* g2_ad     = (const float*)d_in[9];
  const float* g2_bias   = (const float*)d_in[10];
  const float* w_ih_f    = (const float*)d_in[11];
  const float* w_hh_f    = (const float*)d_in[12];
  const float* b_f       = (const float*)d_in[13];
  const float* w_ih_b    = (const float*)d_in[14];
  const float* w_hh_b    = (const float*)d_in[15];
  const float* b_b       = (const float*)d_in[16];
  const float* attn_nw   = (const float*)d_in[17];
  const float* attn_nb   = (const float*)d_in[18];
  const float* attn_sw   = (const float*)d_in[19];
  const float* attn_sb   = (const float*)d_in[20];
  const float* fc_node_W = (const float*)d_in[21];
  const float* fc_node_b = (const float*)d_in[22];
  const float* fc_sp_W   = (const float*)d_in[23];
  const float* fc_sp_b   = (const float*)d_in[24];
  const int*   edge_idx  = (const int*)d_in[25];
  const int*   seq       = (const int*)d_in[26];
  const int*   lengths   = (const int*)d_in[27];
  float* out = (float*)d_out;

  char* ws = (char*)d_ws; size_t off = 0;
  auto alloc = [&](size_t bytes) -> void* {
    void* p = ws + off; off += (bytes + 255) & ~(size_t)255; return p;
  };

  _Float16* feat16 = (_Float16*)alloc((size_t)N_NODES * K1PAD * 2);
  _Float16* W1t    = (_Float16*)alloc((size_t)HC1 * K1PAD * 2);
  float*    H1     = (float*)   alloc((size_t)N_NODES * HC1 * 4);
  float*    as1    = (float*)   alloc((size_t)N_NODES * HEADS * 4);
  float*    ad1    = (float*)   alloc((size_t)N_NODES * HEADS * 4);
  float*    w1     = (float*)   alloc((size_t)ETOT * HEADS * 4);
  float*    den1   = (float*)   alloc((size_t)N_NODES * HEADS * 4);
  float*    z1     = (float*)   alloc((size_t)N_NODES * HC1 * 4);
  _Float16* z1h    = (_Float16*)alloc((size_t)N_NODES * HC1 * 2);
  _Float16* W2t    = (_Float16*)alloc((size_t)EMB * HC1 * 2);
  float*    H2     = (float*)   alloc((size_t)N_NODES * EMB * 4);
  float*    as2    = (float*)   alloc((size_t)N_NODES * 4);
  float*    ad2    = (float*)   alloc((size_t)N_NODES * 4);
  float*    w2     = (float*)   alloc((size_t)ETOT * 4);
  float*    den2   = (float*)   alloc((size_t)N_NODES * 4);
  float*    z2     = (float*)   alloc((size_t)N_NODES * EMB * 4);
  _Float16* wihF   = (_Float16*)alloc((size_t)512 * 128 * 2);
  _Float16* whhF   = (_Float16*)alloc((size_t)512 * 128 * 2);
  _Float16* wihB   = (_Float16*)alloc((size_t)512 * 128 * 2);
  _Float16* whhB   = (_Float16*)alloc((size_t)512 * 128 * 2);
  _Float16* semb   = (_Float16*)alloc((size_t)BT * EMB * 2);
  _Float16* xrev   = (_Float16*)alloc((size_t)BT * EMB * 2);
  float*    XgF    = (float*)   alloc((size_t)BT * 512 * 4);
  float*    XgB    = (float*)   alloc((size_t)BT * 512 * 4);
  _Float16* hFa    = (_Float16*)alloc((size_t)BSZ * LH * 2);
  _Float16* hFb    = (_Float16*)alloc((size_t)BSZ * LH * 2);
  float*    cF     = (float*)   alloc((size_t)BSZ * LH * 4);
  _Float16* hBa    = (_Float16*)alloc((size_t)BSZ * LH * 2);
  _Float16* hBb    = (_Float16*)alloc((size_t)BSZ * LH * 2);
  float*    cB     = (float*)   alloc((size_t)BSZ * LH * 4);
  float*    outF   = (float*)   alloc((size_t)BT * LH * 4);
  float*    outR   = (float*)   alloc((size_t)BT * LH * 4);
  _Float16* ctxn   = (_Float16*)alloc((size_t)BSZ * 256 * 2);
  _Float16* ctxs   = (_Float16*)alloc((size_t)BSZ * 256 * 2);
  _Float16* fcnWt  = (_Float16*)alloc((size_t)N_NODES * 256 * 2);
  _Float16* fcsWt  = (_Float16*)alloc((size_t)NSP * 256 * 2);

  const int TH = 256;
  auto gs = [&](long long n) { return (unsigned)((n + TH - 1) / TH); };

  // ---- weight / feature conversion to f16 (transposed for B-fragment loads)
  k_prep_feat<<<gs((long long)N_NODES * K1PAD), TH, 0, stream>>>(x_coords, node_emb, temporal, feat16);
  k_transpose_cvt<<<gs((long long)HC1 * K1PAD), TH, 0, stream>>>(gat1_W, W1t, 134, HC1, K1PAD);
  k_transpose_cvt<<<gs((long long)EMB * HC1), TH, 0, stream>>>(gat2_W, W2t, HC1, EMB, HC1);
  k_transpose_cvt<<<gs((long long)N_NODES * 256), TH, 0, stream>>>(fc_node_W, fcnWt, 256, N_NODES, 256);
  k_transpose_cvt<<<gs((long long)NSP * 256), TH, 0, stream>>>(fc_sp_W, fcsWt, 256, NSP, 256);
  k_cvt<<<gs(512 * 128), TH, 0, stream>>>(w_ih_f, wihF, 512 * 128);
  k_cvt<<<gs(512 * 128), TH, 0, stream>>>(w_hh_f, whhF, 512 * 128);
  k_cvt<<<gs(512 * 128), TH, 0, stream>>>(w_ih_b, wihB, 512 * 128);
  k_cvt<<<gs(512 * 128), TH, 0, stream>>>(w_hh_b, whhB, 512 * 128);

  // ---- GAT layer 1: [20000,160] x [160,256]  (MB2 x NB4 macro-tiles)
  { int Mg = (N_NODES / 16) / 2, Ng = (HC1 / 16) / 4, waves = Mg * Ng;
    gemm_wmma_blk<2, 4><<<(waves + 3) / 4, 128, 0, stream>>>(
        feat16, K1PAD, W1t, K1PAD, H1, HC1, nullptr, Mg, Ng, K1PAD / 32); }
  k_asad<<<gs((long long)N_NODES * HEADS), TH, 0, stream>>>(H1, g1_as, g1_ad, as1, ad1, HEADS, GH);
  hipMemsetAsync(den1, 0, (size_t)N_NODES * HEADS * 4, stream);
  hipMemsetAsync(z1, 0, (size_t)N_NODES * HC1 * 4, stream);
  k_edge_w<<<gs((long long)ETOT * HEADS), TH, 0, stream>>>(as1, ad1, edge_idx, w1, den1, HEADS);
  k_edge_agg<<<gs((long long)ETOT * HC1), TH, 0, stream>>>(H1, w1, den1, edge_idx, z1, HEADS, GH);
  k_bias_elu_half<<<gs((long long)N_NODES * HC1), TH, 0, stream>>>(z1, g1_bias, z1h, HC1);

  // ---- GAT layer 2: [20000,256] x [256,128]
  { int Mg = (N_NODES / 16) / 2, Ng = (EMB / 16) / 4, waves = Mg * Ng;
    gemm_wmma_blk<2, 4><<<(waves + 3) / 4, 128, 0, stream>>>(
        z1h, HC1, W2t, HC1, H2, EMB, nullptr, Mg, Ng, HC1 / 32); }
  k_asad<<<gs((long long)N_NODES), TH, 0, stream>>>(H2, g2_as, g2_ad, as2, ad2, 1, EMB);
  hipMemsetAsync(den2, 0, (size_t)N_NODES * 4, stream);
  hipMemsetAsync(z2, 0, (size_t)N_NODES * EMB * 4, stream);
  k_edge_w<<<gs((long long)ETOT), TH, 0, stream>>>(as2, ad2, edge_idx, w2, den2, 1);
  k_edge_agg<<<gs((long long)ETOT * EMB), TH, 0, stream>>>(H2, w2, den2, edge_idx, z2, 1, EMB);

  // ---- sequence gather (+bias) in forward & length-reversed order
  k_gather<<<gs((long long)BT * EMB), TH, 0, stream>>>(z2, g2_bias, seq, lengths, semb, xrev);

  // ---- time-parallel LSTM input projections: Xg = x @ W_ih^T  [25600,128]x[128,512]
  { int Mg = (BT / 16) / 2, Ng = (512 / 16) / 4, waves = Mg * Ng;
    gemm_wmma_blk<2, 4><<<(waves + 3) / 4, 128, 0, stream>>>(
        semb, EMB, wihF, EMB, XgF, 512, nullptr, Mg, Ng, EMB / 32);
    gemm_wmma_blk<2, 4><<<(waves + 3) / 4, 128, 0, stream>>>(
        xrev, EMB, wihB, EMB, XgB, 512, nullptr, Mg, Ng, EMB / 32); }

  // ---- recurrent part (sequential over T, fused WMMA + gates)
  hipMemsetAsync(hFa, 0, (size_t)BSZ * LH * 2, stream);
  hipMemsetAsync(cF, 0, (size_t)BSZ * LH * 4, stream);
  hipMemsetAsync(hBa, 0, (size_t)BSZ * LH * 2, stream);
  hipMemsetAsync(cB, 0, (size_t)BSZ * LH * 4, stream);
  dim3 lgrid(BSZ / 16, LH / 16);
  for (int t = 0; t < TT; ++t) {
    _Float16* hin  = (t & 1) ? hFb : hFa;
    _Float16* hout = (t & 1) ? hFa : hFb;
    lstm_step<<<lgrid, 128, 0, stream>>>(hin, whhF, XgF, b_f, cF, hout, outF, t);
    _Float16* hinB  = (t & 1) ? hBb : hBa;
    _Float16* houtB = (t & 1) ? hBa : hBb;
    lstm_step<<<lgrid, 128, 0, stream>>>(hinB, whhB, XgB, b_b, cB, houtB, outR, t);
  }

  // ---- masked attention pooling (both pools)
  k_attn<<<BSZ, 256, 0, stream>>>(outF, outR, lengths, attn_nw, attn_nb, attn_sw, attn_sb,
                                  ctxn, ctxs);

  // ---- final FCs straight into d_out (tuple order: node logits, sp logits)
  { int Mg = (BSZ / 16) / 2, Ng = (N_NODES / 16) / 2, waves = Mg * Ng;   // 1250 N-tiles -> NB=2
    gemm_wmma_blk<2, 2><<<(waves + 3) / 4, 128, 0, stream>>>(
        ctxn, 256, fcnWt, 256, out, N_NODES, fc_node_b, Mg, Ng, 256 / 32); }
  { int Mg = (BSZ / 16) / 2, Ng = (NSP / 16) / 4, waves = Mg * Ng;
    gemm_wmma_blk<2, 4><<<(waves + 3) / 4, 128, 0, stream>>>(
        ctxs, 256, fcsWt, 256, out + (size_t)BSZ * N_NODES, NSP, fc_sp_b,
        Mg, Ng, 256 / 32); }
}
